// PointNet2SSGClassification_56538949485044
// MI455X (gfx1250) — compile-verified, matrix operand order checked by reading, not run on hardware
//
#include <hip/hip_runtime.h>

// ---------------------------------------------------------------------------
// CDNA5 (gfx1250) PointNet++ SSG forward.
// All MLP layers run as bf16 WMMA GEMMs (v_wmma_f32_16x16x32_bf16) with f32
// accumulate. Weights are pre-packed into B-fragment layout; activations are
// row-major bf16 with K padded to multiples of 32 (zero-filled) so A fragments
// are two b128 loads per lane per K-tile. The GEMM computes an MWxNW grid of
// 16x16 tiles per wave (up to 32x64); the K loop is 2x unrolled with ping/pong
// fragment buffers so loads overlap WMMA issue without register copies.
// ---------------------------------------------------------------------------

typedef __attribute__((ext_vector_type(16))) __bf16        v16bf;
typedef __attribute__((ext_vector_type(8)))  float         v8f;
typedef __attribute__((ext_vector_type(4)))  unsigned int  u32x4;

__device__ __forceinline__ unsigned short f2bf(float f) {
  union { float f; unsigned int u; } v; v.f = f;
  return (unsigned short)((v.u + 0x7FFFu + ((v.u >> 16) & 1u)) >> 16);
}
__device__ __forceinline__ float bf2f(unsigned short h) {
  union { float f; unsigned int u; } v; v.u = ((unsigned int)h) << 16;
  return v.f;
}

union FragU { u32x4 q[2]; v16bf v; };

// A fragment: 16x32 bf16, row-major source [lda], ISA layout:
// lane l: h=l>>4, r=l&15, VGPR v holds K pair (k0 + 8h + 2v) for v<4,
// (k0 + 16 + 8h + 2(v-4)) for v>=4  => two contiguous 16B loads.
__device__ __forceinline__ v16bf load_a_frag(const unsigned short* __restrict__ A,
                                             int lda, int m0, int k0, int lane) {
  int h = (lane >> 4) & 1, r = lane & 15;
  const unsigned short* p = A + (size_t)(m0 + r) * lda + k0 + 8 * h;
  FragU u;
  u.q[0] = *(const u32x4*)(p);
  u.q[1] = *(const u32x4*)(p + 16);
  return u.v;
}

// B fragment: pre-packed, tile-contiguous: Wp[((kt*Ntiles+nt)*32+lane)*8 + v]
__device__ __forceinline__ v16bf load_b_frag(const unsigned int* __restrict__ Wp,
                                             int Ntiles, int kt, int nt, int lane) {
  const unsigned int* p = Wp + (((size_t)kt * Ntiles + nt) * 32 + lane) * 8;
  FragU u;
  u.q[0] = *(const u32x4*)(p);
  u.q[1] = *(const u32x4*)(p + 4);
  return u.v;
}

// ---------------------------------------------------------------------------
__global__ void transpose_xyz_kernel(const float* __restrict__ in, float* __restrict__ xyz,
                                     int N, int total) {
  int t = blockIdx.x * blockDim.x + threadIdx.x;
  if (t >= total) return;
  int b = t / N, i = t - b * N;
  const float* src = in + (size_t)b * 3 * N + i;
  float* dst = xyz + (size_t)t * 3;
  dst[0] = src[0]; dst[1] = src[N]; dst[2] = src[2 * N];
}

// Pack W (Kin x Nin f32, row-major) into B-fragment layout, zero-padded.
__global__ void pack_w_kernel(const float* __restrict__ W, unsigned int* __restrict__ Wp,
                              int Kin, int Nin, int Ntiles, int total) {
  int t = blockIdx.x * blockDim.x + threadIdx.x;
  if (t >= total) return;                 // total = Ktiles*Ntiles*32
  int lane = t & 31;
  int tile = t >> 5;
  int nt = tile % Ntiles, kt = tile / Ntiles;
  int h = (lane >> 4) & 1, c = lane & 15;
  int n = nt * 16 + c;
  unsigned int out[8];
#pragma unroll
  for (int v = 0; v < 8; ++v) {
    int k = kt * 32 + 8 * h + ((v < 4) ? (2 * v) : (16 + 2 * (v - 4)));
    float f0 = (k < Kin && n < Nin) ? W[(size_t)k * Nin + n] : 0.0f;
    float f1 = ((k + 1) < Kin && n < Nin) ? W[(size_t)(k + 1) * Nin + n] : 0.0f;
    out[v] = (unsigned int)f2bf(f0) | ((unsigned int)f2bf(f1) << 16);
  }
  unsigned int* p = Wp + (size_t)t * 8;
#pragma unroll
  for (int v = 0; v < 8; ++v) p[v] = out[v];
}

__global__ void pack_bias_kernel(const float* __restrict__ b, float* __restrict__ out,
                                 int Nin, int Npad) {
  int t = blockIdx.x * blockDim.x + threadIdx.x;
  if (t < Npad) out[t] = (t < Nin) ? b[t] : 0.0f;
}

// ---------------------------------------------------------------------------
// GEMM: C = act(A[M,Kpad] * Wpacked + bias). MWxNW 16x16 tiles per wave.
// K loop 2x unrolled with ping/pong fragments (no inter-buffer copies).
template <int MW, int NW>
__global__ void __launch_bounds__(256)
gemm_tile_kernel(const unsigned short* __restrict__ A, const unsigned int* __restrict__ Bp,
                 const float* __restrict__ bias, unsigned short* __restrict__ Obf,
                 float* __restrict__ Of32, int MtW, int NtW, int Ntiles, int Kpad,
                 int Nact, int ldo, int relu) {
  int wave = (int)((blockIdx.x * blockDim.x + threadIdx.x) >> 5);
  int lane = threadIdx.x & 31;
  if (wave >= MtW * NtW) return;
  int ntw = wave % NtW, mtw = wave / NtW;
  int m0 = mtw * MW * 16;
  int nt0 = ntw * NW;
  int Npad = Ntiles * 16;
  int Ktiles = Kpad >> 5;

  v8f acc[MW][NW];
#pragma unroll
  for (int mi = 0; mi < MW; ++mi)
#pragma unroll
    for (int ni = 0; ni < NW; ++ni) { v8f z = {}; acc[mi][ni] = z; }

  v16bf a0[MW], b0[NW], a1[MW], b1[NW];
#pragma unroll
  for (int mi = 0; mi < MW; ++mi) a0[mi] = load_a_frag(A, Kpad, m0 + 16 * mi, 0, lane);
#pragma unroll
  for (int ni = 0; ni < NW; ++ni) b0[ni] = load_b_frag(Bp, Ntiles, 0, nt0 + ni, lane);

  int kt = 0;
  while (true) {
    // ---- stage 0: compute on buf0 (K-tile kt), load buf1 (kt+1) ----
    if (kt + 1 < Ktiles) {
#pragma unroll
      for (int mi = 0; mi < MW; ++mi)
        a1[mi] = load_a_frag(A, Kpad, m0 + 16 * mi, (kt + 1) * 32, lane);
#pragma unroll
      for (int ni = 0; ni < NW; ++ni)
        b1[ni] = load_b_frag(Bp, Ntiles, kt + 1, nt0 + ni, lane);
    }
#pragma unroll
    for (int mi = 0; mi < MW; ++mi)
#pragma unroll
      for (int ni = 0; ni < NW; ++ni)
        acc[mi][ni] = __builtin_amdgcn_wmma_f32_16x16x32_bf16(
            false, a0[mi], false, b0[ni], (short)0, acc[mi][ni], false, false);
    if (++kt >= Ktiles) break;

    // ---- stage 1: compute on buf1 (K-tile kt), load buf0 (kt+1) ----
    if (kt + 1 < Ktiles) {
#pragma unroll
      for (int mi = 0; mi < MW; ++mi)
        a0[mi] = load_a_frag(A, Kpad, m0 + 16 * mi, (kt + 1) * 32, lane);
#pragma unroll
      for (int ni = 0; ni < NW; ++ni)
        b0[ni] = load_b_frag(Bp, Ntiles, kt + 1, nt0 + ni, lane);
    }
#pragma unroll
    for (int mi = 0; mi < MW; ++mi)
#pragma unroll
      for (int ni = 0; ni < NW; ++ni)
        acc[mi][ni] = __builtin_amdgcn_wmma_f32_16x16x32_bf16(
            false, a1[mi], false, b1[ni], (short)0, acc[mi][ni], false, false);
    if (++kt >= Ktiles) break;
  }

  int h = (lane >> 4) & 1, c = lane & 15;
#pragma unroll
  for (int ni = 0; ni < NW; ++ni) {
    int col = (nt0 + ni) * 16 + c;
    float bv = bias ? bias[col] : 0.0f;
#pragma unroll
    for (int mi = 0; mi < MW; ++mi) {
      float vals[8];
#pragma unroll
      for (int r = 0; r < 8; ++r) {
        float x = acc[mi][ni][r] + bv;
        vals[r] = relu ? fmaxf(x, 0.0f) : x;
      }
      int rbase = m0 + 16 * mi + 8 * h;
      if (Obf) {
#pragma unroll
        for (int r = 0; r < 8; ++r)
          Obf[(size_t)(rbase + r) * Npad + col] = f2bf(vals[r]);
      }
      if (Of32) {
#pragma unroll
        for (int r = 0; r < 8; ++r)
          if (col < Nact) Of32[(size_t)(rbase + r) * ldo + col] = vals[r];
      }
    }
  }
}

// ---------------------------------------------------------------------------
// Farthest point sampling: one block per batch, serial npoint loop.
__global__ void fps_kernel(const float* __restrict__ xyz, int N, int npoint,
                           float* __restrict__ dist, int* __restrict__ fidx) {
  int b = blockIdx.x, tid = threadIdx.x;
  const float* X = xyz + (size_t)b * N * 3;
  float* D = dist + (size_t)b * N;
  int* F = fidx + (size_t)b * npoint;
  __shared__ float sval[256];
  __shared__ int   sidx[256];
  __shared__ int   sfar;
  for (int i = tid; i < N; i += 256) D[i] = 1e10f;
  if (tid == 0) sfar = 0;
  __syncthreads();
  for (int it = 0; it < npoint; ++it) {
    int far = sfar;
    if (tid == 0) F[it] = far;
    float cx = X[far * 3], cy = X[far * 3 + 1], cz = X[far * 3 + 2];
    float best = -1.0f; int besti = 0x7fffffff;
    for (int i = tid; i < N; i += 256) {
      float dx = X[i * 3] - cx, dy = X[i * 3 + 1] - cy, dz = X[i * 3 + 2] - cz;
      float d = fminf(D[i], dx * dx + dy * dy + dz * dz);
      D[i] = d;
      if (d > best) { best = d; besti = i; }
    }
    sval[tid] = best; sidx[tid] = besti;
    __syncthreads();
    for (int s = 128; s > 0; s >>= 1) {
      if (tid < s) {
        if (sval[tid + s] > sval[tid] ||
            (sval[tid + s] == sval[tid] && sidx[tid + s] < sidx[tid])) {
          sval[tid] = sval[tid + s]; sidx[tid] = sidx[tid + s];
        }
      }
      __syncthreads();
    }
    if (tid == 0) sfar = sidx[0];
    __syncthreads();
  }
}

__global__ void gather_xyz_kernel(const float* __restrict__ xyz, const int* __restrict__ fidx,
                                  float* __restrict__ out, int N, int npoint, int total) {
  int t = blockIdx.x * blockDim.x + threadIdx.x;
  if (t >= total) return;
  int b = t / npoint;
  int j = fidx[t];
  const float* s = xyz + ((size_t)b * N + j) * 3;
  float* d = out + (size_t)t * 3;
  d[0] = s[0]; d[1] = s[1]; d[2] = s[2];
}

// First nsample in-range indices (== reference's index-sorted selection), pad w/ first.
__global__ void ball_query_kernel(const float* __restrict__ xyz, const float* __restrict__ nxyz,
                                  int* __restrict__ out, int N, int npoint, int nsample,
                                  float r2, int total) {
  int t = blockIdx.x * blockDim.x + threadIdx.x;
  if (t >= total) return;
  int b = t / npoint;
  const float* X = xyz + (size_t)b * N * 3;
  float cx = nxyz[t * 3], cy = nxyz[t * 3 + 1], cz = nxyz[t * 3 + 2];
  int* O = out + (size_t)t * nsample;
  int cnt = 0, first = 0;
  for (int j = 0; j < N && cnt < nsample; ++j) {
    float dx = X[j * 3] - cx, dy = X[j * 3 + 1] - cy, dz = X[j * 3 + 2] - cz;
    if (dx * dx + dy * dy + dz * dz <= r2) {
      if (cnt == 0) first = j;
      O[cnt++] = j;
    }
  }
  for (; cnt < nsample; ++cnt) O[cnt] = first;
}

// SA1 grouping: rows = relative xyz (3), zero-pad to Kpad, bf16.
__global__ void group_rel_kernel(const float* __restrict__ xyz, const float* __restrict__ nxyz,
                                 const int* __restrict__ ball, unsigned short* __restrict__ g,
                                 int N, int npoint, int nsample, int Kpad, int total) {
  int t = blockIdx.x * blockDim.x + threadIdx.x;
  if (t >= total) return;
  int s = (t / nsample) % npoint;
  int b = t / (nsample * npoint);
  int j = ball[t];
  const float* X = xyz + ((size_t)b * N + j) * 3;
  const float* C = nxyz + ((size_t)b * npoint + s) * 3;
  unsigned short* row = g + (size_t)t * Kpad;
  row[0] = f2bf(X[0] - C[0]); row[1] = f2bf(X[1] - C[1]); row[2] = f2bf(X[2] - C[2]);
  for (int c = 3; c < Kpad; ++c) row[c] = 0;
}

// SA2 grouping: [rel xyz (3) | feats (C)] zero-padded to Kpad.
__global__ void group_feat_kernel(const float* __restrict__ xyz, const float* __restrict__ nxyz,
                                  const int* __restrict__ ball, const unsigned short* __restrict__ feats,
                                  int C, unsigned short* __restrict__ g,
                                  int N, int npoint, int nsample, int Kpad, int total) {
  int t = blockIdx.x * blockDim.x + threadIdx.x;
  if (t >= total) return;
  int s = (t / nsample) % npoint;
  int b = t / (nsample * npoint);
  int j = ball[t];
  const float* X = xyz + ((size_t)b * N + j) * 3;
  const float* Cc = nxyz + ((size_t)b * npoint + s) * 3;
  const unsigned short* F = feats + ((size_t)b * N + j) * C;
  unsigned short* row = g + (size_t)t * Kpad;
  row[0] = f2bf(X[0] - Cc[0]); row[1] = f2bf(X[1] - Cc[1]); row[2] = f2bf(X[2] - Cc[2]);
  for (int i = 0; i < C; ++i) row[3 + i] = F[i];
  for (int c = 3 + C; c < Kpad; ++c) row[c] = 0;
}

// SA3 global grouping: [xyz (3) | feats (C)] zero-padded.
__global__ void group_global_kernel(const float* __restrict__ xyz, const unsigned short* __restrict__ feats,
                                    int C, unsigned short* __restrict__ g, int Kpad, int total) {
  int t = blockIdx.x * blockDim.x + threadIdx.x;
  if (t >= total) return;
  const float* X = xyz + (size_t)t * 3;
  const unsigned short* F = feats + (size_t)t * C;
  unsigned short* row = g + (size_t)t * Kpad;
  row[0] = f2bf(X[0]); row[1] = f2bf(X[1]); row[2] = f2bf(X[2]);
  for (int i = 0; i < C; ++i) row[3 + i] = F[i];
  for (int c = 3 + C; c < Kpad; ++c) row[c] = 0;
}

// Max over gs consecutive rows (post-ReLU activations).
__global__ void maxpool_kernel(const unsigned short* __restrict__ in, unsigned short* __restrict__ out,
                               int gs, int C, int groups) {
  int g = blockIdx.x;
  if (g >= groups) return;
  for (int c = threadIdx.x; c < C; c += blockDim.x) {
    const unsigned short* p = in + (size_t)g * gs * C + c;
    float m = -3.4e38f;
    for (int r = 0; r < gs; ++r) m = fmaxf(m, bf2f(p[(size_t)r * C]));
    out[(size_t)g * C + c] = f2bf(m);
  }
}

// ---------------------------------------------------------------------------
extern "C" void kernel_launch(void* const* d_in, const int* in_sizes, int n_in,
                              void* d_out, int out_size, void* d_ws, size_t ws_size,
                              hipStream_t stream) {
  (void)in_sizes; (void)n_in; (void)out_size; (void)ws_size;
  const int B = 16, N = 4096;
  const int NP1 = 512, NS1 = 32, NP2 = 128, NS2 = 64;

  char* base = (char*)d_ws;
  size_t off = 0;
  auto alloc = [&](size_t bytes) -> void* {
    off = (off + 255) & ~(size_t)255;
    void* p = base + off;
    off += bytes;
    return p;
  };

  float* xyz   = (float*)alloc((size_t)B * N * 3 * 4);
  float* dist  = (float*)alloc((size_t)B * N * 4);
  int*   fidx1 = (int*)alloc((size_t)B * NP1 * 4);
  float* l1xyz = (float*)alloc((size_t)B * NP1 * 3 * 4);
  int*   ball1 = (int*)alloc((size_t)B * NP1 * NS1 * 4);
  unsigned short* l1f = (unsigned short*)alloc((size_t)B * NP1 * 128 * 2);
  int*   fidx2 = (int*)alloc((size_t)B * NP2 * 4);
  float* l2xyz = (float*)alloc((size_t)B * NP2 * 3 * 4);
  int*   ball2 = (int*)alloc((size_t)B * NP2 * NS2 * 4);
  unsigned short* l2f = (unsigned short*)alloc((size_t)B * NP2 * 256 * 2);
  unsigned short* l3f = (unsigned short*)alloc((size_t)B * 1024 * 2);

  // Layer metadata: 12 layers, K padded to x32, N padded to x16.
  const int LK[12]  = {3, 64, 64,  131, 128, 128,  259, 256, 512,  1024, 512, 256};
  const int LN[12]  = {64, 64, 128, 128, 128, 256,  256, 512, 1024, 512,  256, 40};
  const int LKp[12] = {32, 64, 64,  160, 128, 128,  288, 256, 512,  1024, 512, 256};
  const int LNp[12] = {64, 64, 128, 128, 128, 256,  256, 512, 1024, 512,  256, 48};
  unsigned int* Wp[12]; float* Bb[12];
  for (int l = 0; l < 12; ++l) {
    Wp[l] = (unsigned int*)alloc((size_t)LKp[l] * LNp[l] * 2);
    Bb[l] = (float*)alloc((size_t)LNp[l] * 4);
  }
  // Ping-pong activation buffers (peak: g2=40MB on A, SA1/SA2 h3=64MB on B).
  unsigned short* bufA = (unsigned short*)alloc((size_t)131072 * 160 * 2);
  unsigned short* bufB = (unsigned short*)alloc((size_t)262144 * 128 * 2);

#define CEILDIV(a, b) (((a) + (b) - 1) / (b))

  // d_in layout (setup_inputs dict order): [0]=inputs, then (W,b) x12.
  const float* in0 = (const float*)d_in[0];
  transpose_xyz_kernel<<<CEILDIV(B * N, 256), 256, 0, stream>>>(in0, xyz, N, B * N);

  for (int l = 0; l < 12; ++l) {
    const float* W  = (const float*)d_in[1 + 2 * l];
    const float* bv = (const float*)d_in[2 + 2 * l];
    int Kt = LKp[l] / 32, Nt = LNp[l] / 16;
    int tot = Kt * Nt * 32;
    pack_w_kernel<<<CEILDIV(tot, 256), 256, 0, stream>>>(W, Wp[l], LK[l], LN[l], Nt, tot);
    pack_bias_kernel<<<CEILDIV(LNp[l], 256), 256, 0, stream>>>(bv, Bb[l], LN[l], LNp[l]);
  }

  auto gemm = [&](const unsigned short* A, int l, unsigned short* Obf, float* Of32,
                  int M, int relu, int Nact, int ldo) {
    int Mt = M / 16, Nt = LNp[l] / 16;
    int MW = (Mt % 2 == 0) ? 2 : 1;
    int NW = (Nt % 4 == 0) ? 4 : ((Nt % 2 == 0) ? 2 : 1);
    int waves = (Mt / MW) * (Nt / NW);
    dim3 grid(CEILDIV(waves, 8));
#define GLAUNCH(MW_, NW_)                                                      \
    gemm_tile_kernel<MW_, NW_><<<grid, 256, 0, stream>>>(                      \
        A, Wp[l], Bb[l], Obf, Of32, Mt / MW_, Nt / NW_, Nt, LKp[l], Nact, ldo, relu)
    if      (MW == 2 && NW == 4) GLAUNCH(2, 4);
    else if (MW == 2 && NW == 2) GLAUNCH(2, 2);
    else if (MW == 2 && NW == 1) GLAUNCH(2, 1);
    else if (MW == 1 && NW == 4) GLAUNCH(1, 4);
    else if (MW == 1 && NW == 2) GLAUNCH(1, 2);
    else                         GLAUNCH(1, 1);
#undef GLAUNCH
  };

  // ---- SA1 ----
  fps_kernel<<<B, 256, 0, stream>>>(xyz, N, NP1, dist, fidx1);
  gather_xyz_kernel<<<CEILDIV(B * NP1, 256), 256, 0, stream>>>(xyz, fidx1, l1xyz, N, NP1, B * NP1);
  ball_query_kernel<<<CEILDIV(B * NP1, 128), 128, 0, stream>>>(xyz, l1xyz, ball1, N, NP1, NS1, 0.04f, B * NP1);
  const int R1 = B * NP1 * NS1;  // 262144 rows
  group_rel_kernel<<<CEILDIV(R1, 256), 256, 0, stream>>>(xyz, l1xyz, ball1, bufA, N, NP1, NS1, 32, R1);
  gemm(bufA, 0, bufB, nullptr, R1, 1, 0, 0);
  gemm(bufB, 1, bufA, nullptr, R1, 1, 0, 0);
  gemm(bufA, 2, bufB, nullptr, R1, 1, 0, 0);
  maxpool_kernel<<<B * NP1, 128, 0, stream>>>(bufB, l1f, NS1, 128, B * NP1);

  // ---- SA2 ----
  fps_kernel<<<B, 256, 0, stream>>>(l1xyz, NP1, NP2, dist, fidx2);
  gather_xyz_kernel<<<CEILDIV(B * NP2, 256), 256, 0, stream>>>(l1xyz, fidx2, l2xyz, NP1, NP2, B * NP2);
  ball_query_kernel<<<CEILDIV(B * NP2, 128), 128, 0, stream>>>(l1xyz, l2xyz, ball2, NP1, NP2, NS2, 0.16f, B * NP2);
  const int R2 = B * NP2 * NS2;  // 131072 rows
  group_feat_kernel<<<CEILDIV(R2, 256), 256, 0, stream>>>(l1xyz, l2xyz, ball2, l1f, 128, bufA, NP1, NP2, NS2, 160, R2);
  gemm(bufA, 3, bufB, nullptr, R2, 1, 0, 0);
  gemm(bufB, 4, bufA, nullptr, R2, 1, 0, 0);
  gemm(bufA, 5, bufB, nullptr, R2, 1, 0, 0);
  maxpool_kernel<<<B * NP2, 256, 0, stream>>>(bufB, l2f, NS2, 256, B * NP2);

  // ---- SA3 (global) ----
  const int R3 = B * NP2;  // 2048 rows
  group_global_kernel<<<CEILDIV(R3, 256), 256, 0, stream>>>(l2xyz, l2f, 256, bufA, 288, R3);
  gemm(bufA, 6, bufB, nullptr, R3, 1, 0, 0);
  gemm(bufB, 7, bufA, nullptr, R3, 1, 0, 0);
  gemm(bufA, 8, bufB, nullptr, R3, 1, 0, 0);
  maxpool_kernel<<<B, 256, 0, stream>>>(bufB, l3f, NP2, 1024, B);

  // ---- Decoder (M = 16 = one WMMA M-tile) ----
  gemm(l3f, 9, bufA, nullptr, B, 1, 0, 0);
  gemm(bufA, 10, bufB, nullptr, B, 1, 0, 0);
  gemm(bufB, 11, nullptr, (float*)d_out, B, 0, 40, 40);  // logits, no ReLU
#undef CEILDIV
}
